// GCN_12695923327069
// MI455X (gfx1250) — matrix operand order repaired and checked by reading, not compiled, not run on hardware
//
#include <hip/hip_runtime.h>
#include <cstdint>
#include <cstddef>

// ---------------------------------------------------------------------------
// CDNA5 (gfx1250) GCN pipeline: scatter-aggregate (atomic f32) + bf16 WMMA GEMM
// All GEMM operands padded (M->16, K->32) so the WMMA inner loop is branch-free.
// ---------------------------------------------------------------------------

typedef __attribute__((ext_vector_type(16))) __bf16 v16bf;
typedef __attribute__((ext_vector_type(8)))  float  v8f;

// ------------------------- small utility kernels ---------------------------

__global__ void fill_f32_kernel(float* __restrict__ p, long long n, float v) {
  long long i = (long long)blockIdx.x * blockDim.x + threadIdx.x;
  long long stride = (long long)gridDim.x * blockDim.x;
  for (; i < n; i += stride) p[i] = v;
}

// init h_g region of d_out: [G, 2H]; cols [0,H)=0 (sum), cols [H,2H)=-inf (max)
__global__ void init_hg_kernel(unsigned* __restrict__ hg, int G, int H) {
  long long total = (long long)G * 2 * H;
  long long i = (long long)blockIdx.x * blockDim.x + threadIdx.x;
  long long stride = (long long)gridDim.x * blockDim.x;
  for (; i < total; i += stride) {
    int c = (int)(i % (2 * H));
    hg[i] = (c < H) ? 0u : 0xFF800000u;  // -inf bits
  }
}

__global__ void degree_kernel(const int* __restrict__ src, const int* __restrict__ dst,
                              float* __restrict__ outdeg, float* __restrict__ indeg, int E) {
  int e = blockIdx.x * blockDim.x + threadIdx.x;
  if (e >= E) return;
  atomicAdd(&outdeg[src[e]], 1.0f);
  atomicAdd(&indeg[dst[e]], 1.0f);
}

__global__ void norm_kernel(const float* __restrict__ outdeg, const float* __restrict__ indeg,
                            float* __restrict__ onorm, float* __restrict__ inorm, int N) {
  int i = blockIdx.x * blockDim.x + threadIdx.x;
  if (i >= N) return;
  onorm[i] = rsqrtf(fmaxf(outdeg[i], 1.0f));
  inorm[i] = rsqrtf(fmaxf(indeg[i], 1.0f));
}

// W[K,N] f32 -> Wt[N,Kpad] bf16 (transposed, K zero-padded to Kpad)
__global__ void convert_transpose_bf16_kernel(const float* __restrict__ W,
                                              __bf16* __restrict__ Wt,
                                              int K, int Kpad, int N) {
  long long idx = (long long)blockIdx.x * blockDim.x + threadIdx.x;
  if (idx >= (long long)N * Kpad) return;
  int n = (int)(idx / Kpad);
  int k = (int)(idx % Kpad);
  Wt[idx] = (k < K) ? (__bf16)W[(size_t)k * N + n] : (__bf16)0.0f;
}

// agg[dst] += x[src] * out_norm[src], one thread per (edge, 16-float chunk)
__global__ void edge_scatter_kernel(const float* __restrict__ x, const float* __restrict__ onorm,
                                    const int* __restrict__ src, const int* __restrict__ dst,
                                    float* __restrict__ agg, int E, int F, int aggStride) {
  const int chunks = F >> 4;  // 16 floats per thread
  long long idx = (long long)blockIdx.x * blockDim.x + threadIdx.x;
  long long total = (long long)E * chunks;
  if (idx >= total) return;
  const int e  = (int)(idx / chunks);
  const int f0 = (int)(idx % chunks) << 4;
  const int s = src[e], d = dst[e];
  const float sc = onorm[s];
  const float4* xs = (const float4*)(x + (size_t)s * F + f0);
  float* o = agg + (size_t)d * aggStride + f0;
#pragma unroll
  for (int j = 0; j < 4; ++j) {
    const float4 v = xs[j];
    atomicAdd(o + 4 * j + 0, v.x * sc);
    atomicAdd(o + 4 * j + 1, v.y * sc);
    atomicAdd(o + 4 * j + 2, v.z * sc);
    atomicAdd(o + 4 * j + 3, v.w * sc);
  }
}

// ------------------------- WMMA GEMM (the compute core) --------------------
// C[Mpad,N] = (rowScale[m] * A[m,:]) @ W + bias
//   A: f32 [Mpad, K]  (M padded to 16, K padded to 32 -> NO guards anywhere)
//   W: pre-transposed bf16 [N, K]
// Wave computes a 16(M) x 64(N) strip; K stepped by 32 (v_wmma_f32_16x16x32_bf16).
// Fragment layouts used here (A/B K order contiguous, C/D per ISA 7.12.2; g = lane>>4, r = lane&15):
//   A 16x32 bf16 : row=r; elems 0..15 -> K=16g+e
//   B 32x16 bf16 : col=r; elems 0..15 -> K=16g+e (contiguous in Wt[col*K+..])
//   C/D 16x16 f32: n=r, m=vr+8g for accumulator element vr
__global__ void __launch_bounds__(256)
gemm_rowscale_bf16_kernel(const float* __restrict__ A, const float* __restrict__ rowScale,
                          const __bf16* __restrict__ Wt, const float* __restrict__ bias,
                          float* __restrict__ C, int Mpad, int K, int N) {
  const int lane = threadIdx.x & 31;
  const int wave = threadIdx.x >> 5;
  const int tilesN = N >> 6;                      // number of 64-wide strips
  const int gw = blockIdx.x * 8 + wave;
  if (gw >= (Mpad >> 4) * tilesN) return;         // wave-uniform exit
  const int m0  = (gw / tilesN) << 4;
  const int n64 = (gw % tilesN) << 6;

  const int g = lane >> 4;
  const int r = lane & 15;
  const int row = m0 + r;
  const float sc = rowScale[row];
  const float* __restrict__ Ap0 = A + (size_t)row * K + 16 * g;       // elems 0..7
  const float* __restrict__ Ap1 = Ap0 + 8;                            // elems 8..15
  const __bf16* __restrict__ Bp0 = Wt + (size_t)(n64 + r) * K + 16 * g;

  v8f acc[4] = {};

  for (int k0 = 0; k0 < K; k0 += 32) {
    // ---- A fragment (unguarded: M and K fully padded) ----
    v16bf a;
    {
      float4 p = *(const float4*)(Ap0 + k0);
      float4 q = *(const float4*)(Ap0 + k0 + 4);
      a[0] = (__bf16)(p.x * sc); a[1] = (__bf16)(p.y * sc);
      a[2] = (__bf16)(p.z * sc); a[3] = (__bf16)(p.w * sc);
      a[4] = (__bf16)(q.x * sc); a[5] = (__bf16)(q.y * sc);
      a[6] = (__bf16)(q.z * sc); a[7] = (__bf16)(q.w * sc);
    }
    {
      float4 p = *(const float4*)(Ap1 + k0);
      float4 q = *(const float4*)(Ap1 + k0 + 4);
      a[8]  = (__bf16)(p.x * sc); a[9]  = (__bf16)(p.y * sc);
      a[10] = (__bf16)(p.z * sc); a[11] = (__bf16)(p.w * sc);
      a[12] = (__bf16)(q.x * sc); a[13] = (__bf16)(q.y * sc);
      a[14] = (__bf16)(q.z * sc); a[15] = (__bf16)(q.w * sc);
    }
    // ---- B fragments + WMMA for 4 N-subtiles ----
#pragma unroll
    for (int s = 0; s < 4; ++s) {
      const v16bf b = *(const v16bf*)(Bp0 + (size_t)(s << 4) * K + k0);
      acc[s] = __builtin_amdgcn_wmma_f32_16x16x32_bf16(
          false, a, false, b, (short)0, acc[s], false, false);
    }
  }

  // ---- store D + bias (unguarded: C has Mpad rows) ----
#pragma unroll
  for (int s = 0; s < 4; ++s) {
    const int n = n64 + (s << 4) + r;
    const float bi = bias[n];
    float* __restrict__ Cp = C + (size_t)(m0 + 8 * g) * N + n;
#pragma unroll
    for (int vr = 0; vr < 8; ++vr) Cp[(size_t)vr * N] = acc[s][vr] + bi;
  }
}

// ------------------------- readout + predictor -----------------------------

static __device__ __forceinline__ void atomic_max_f32(float* addr, float val) {
  if (!(val == val)) return;  // NaN: skip
  if (val >= 0.0f) atomicMax((int*)addr, __float_as_int(val));
  else             atomicMin((unsigned int*)addr, (unsigned int)__float_as_int(val));
}

// block per node: w = sigmoid(h.w_atom + b); hg[g, 0:H] += h*w ; hg[g, H:2H] max= h
__global__ void __launch_bounds__(256)
readout_node_kernel(const float* __restrict__ h, const int* __restrict__ gid,
                    const float* __restrict__ w_atom, const float* __restrict__ b_atom,
                    float* __restrict__ hg, int H) {
  __shared__ float red[256];
  const int i = blockIdx.x;
  const int c = threadIdx.x;
  const float hv = h[(size_t)i * H + c];
  red[c] = hv * w_atom[c];
  __syncthreads();
  for (int off = 128; off > 0; off >>= 1) {
    if (c < off) red[c] += red[c + off];
    __syncthreads();
  }
  const float wgt = 1.0f / (1.0f + __expf(-(red[0] + b_atom[0])));
  const int g = gid[i];
  atomicAdd(&hg[(size_t)g * (2 * H) + c], hv * wgt);
  atomic_max_f32(&hg[(size_t)g * (2 * H) + H + c], hv);
}

// replace non-finite max entries (empty graphs -> -inf) with 0
__global__ void fix_max_kernel(float* __restrict__ hg, int G, int H) {
  long long idx = (long long)blockIdx.x * blockDim.x + threadIdx.x;
  if (idx >= (long long)G * H) return;
  int g = (int)(idx / H), c = (int)(idx % H);
  float* p = &hg[(size_t)g * (2 * H) + H + c];
  unsigned u = __float_as_uint(*p);
  if ((u & 0x7F800000u) == 0x7F800000u) *p = 0.0f;
}

// block per graph: pred = relu(hg @ P1 + pb1) @ P2 + pb2
__global__ void __launch_bounds__(256)
predictor_kernel(const float* __restrict__ hg, const float* __restrict__ P1,
                 const float* __restrict__ pb1, const float* __restrict__ P2,
                 const float* __restrict__ pb2, float* __restrict__ pred, int H) {
  __shared__ float row[512];
  __shared__ float red[256];
  const int g = blockIdx.x;
  const int c = threadIdx.x;
  const int H2 = 2 * H;
  row[c]       = hg[(size_t)g * H2 + c];
  row[c + 256] = hg[(size_t)g * H2 + 256 + c];
  __syncthreads();
  float t = pb1[c];
  for (int k = 0; k < H2; ++k) t += row[k] * P1[(size_t)k * H + c];
  t = fmaxf(t, 0.0f);
  red[c] = t * P2[c];
  __syncthreads();
  for (int off = 128; off > 0; off >>= 1) {
    if (c < off) red[c] += red[c + off];
    __syncthreads();
  }
  if (c == 0) pred[g] = red[0] + pb2[0];
}

// ------------------------------- launcher ----------------------------------

extern "C" void kernel_launch(void* const* d_in, const int* in_sizes, int n_in,
                              void* d_out, int out_size, void* d_ws, size_t ws_size,
                              hipStream_t stream) {
  const float* feats  = (const float*)d_in[0];
  const int*   src    = (const int*)d_in[1];
  const int*   dst    = (const int*)d_in[2];
  const int*   gid    = (const int*)d_in[3];
  const float* W1 = (const float*)d_in[4];  const float* b1 = (const float*)d_in[5];
  const float* W2 = (const float*)d_in[6];  const float* b2 = (const float*)d_in[7];
  const float* W3 = (const float*)d_in[8];  const float* b3 = (const float*)d_in[9];
  const float* w_atom = (const float*)d_in[10]; const float* b_atom = (const float*)d_in[11];
  const float* P1 = (const float*)d_in[12]; const float* pb1 = (const float*)d_in[13];
  const float* P2 = (const float*)d_in[14]; const float* pb2 = (const float*)d_in[15];

  const int H  = in_sizes[5];            // 256
  const int F0 = in_sizes[4] / H;        // 16
  const int nNodes = in_sizes[0] / F0;   // 100000
  const int nEdges = in_sizes[1];        // 300000
  const int G = out_size / (2 * H + 1);  // 4000

  const int Mpad = (nNodes + 15) & ~15;  // pad rows to WMMA tile height
  const int K1p  = 32;                   // layer-1 K (16) padded to one WMMA step

  // ---- workspace carve ----
  auto align256 = [](size_t x) { return (x + 255) & ~(size_t)255; };
  char* ws = (char*)d_ws;
  size_t off = 0;
  float* agg    = (float*)(ws + off); off += align256((size_t)Mpad * H * sizeof(float));
  float* hbuf   = (float*)(ws + off); off += align256((size_t)Mpad * H * sizeof(float));
  float* outdeg = (float*)(ws + off); off += align256((size_t)nNodes * sizeof(float));
  float* indeg  = (float*)(ws + off); off += align256((size_t)nNodes * sizeof(float));
  float* onorm  = (float*)(ws + off); off += align256((size_t)nNodes * sizeof(float));
  float* inorm  = (float*)(ws + off); off += align256((size_t)Mpad * sizeof(float));
  __bf16* Wbf   = (__bf16*)(ws + off); off += align256((size_t)H * H * sizeof(__bf16));
  (void)ws_size;

  float* hg   = (float*)d_out;                       // [G, 2H]
  float* pred = (float*)d_out + (size_t)G * 2 * H;   // [G]

  const int T = 256;
  auto blocks = [&](long long n) { return (int)((n + T - 1) / T); };

  // ---- degrees & norms (inorm padded so GEMM row-scale loads are unguarded) ----
  fill_f32_kernel<<<blocks(nNodes), T, 0, stream>>>(outdeg, nNodes, 0.0f);
  fill_f32_kernel<<<blocks(nNodes), T, 0, stream>>>(indeg, nNodes, 0.0f);
  fill_f32_kernel<<<blocks(Mpad), T, 0, stream>>>(inorm, Mpad, 1.0f);
  degree_kernel<<<blocks(nEdges), T, 0, stream>>>(src, dst, outdeg, indeg, nEdges);
  norm_kernel<<<blocks(nNodes), T, 0, stream>>>(outdeg, indeg, onorm, inorm, nNodes);

  const int gemmWaves  = (Mpad >> 4) * (H >> 6);
  const int gemmBlocks = (gemmWaves + 7) / 8;

  // ---- layer 1: feats[nNodes,16], agg row-stride K1p=32, K zero-padded ----
  {
    fill_f32_kernel<<<blocks((long long)Mpad * K1p), T, 0, stream>>>(agg, (long long)Mpad * K1p, 0.0f);
    convert_transpose_bf16_kernel<<<blocks((long long)H * K1p), T, 0, stream>>>(W1, Wbf, F0, K1p, H);
    edge_scatter_kernel<<<blocks((long long)nEdges * (F0 >> 4)), T, 0, stream>>>(
        feats, onorm, src, dst, agg, nEdges, F0, K1p);
    gemm_rowscale_bf16_kernel<<<gemmBlocks, 256, 0, stream>>>(
        agg, inorm, Wbf, b1, hbuf, Mpad, K1p, H);
  }

  // ---- layers 2 & 3: hbuf[nNodes,256] ----
  const float* Ws[2] = {W2, W3};
  const float* bs[2] = {b2, b3};
  for (int l = 0; l < 2; ++l) {
    fill_f32_kernel<<<blocks((long long)Mpad * H), T, 0, stream>>>(agg, (long long)Mpad * H, 0.0f);
    convert_transpose_bf16_kernel<<<blocks((long long)H * H), T, 0, stream>>>(Ws[l], Wbf, H, H, H);
    edge_scatter_kernel<<<blocks((long long)nEdges * (H >> 4)), T, 0, stream>>>(
        hbuf, onorm, src, dst, agg, nEdges, H, H);
    gemm_rowscale_bf16_kernel<<<gemmBlocks, 256, 0, stream>>>(
        agg, inorm, Wbf, bs[l], hbuf, Mpad, H, H);  // writes hbuf, reads agg: safe
  }

  // ---- readout into d_out ----
  init_hg_kernel<<<blocks((long long)G * 2 * H), T, 0, stream>>>((unsigned*)hg, G, H);
  readout_node_kernel<<<nNodes, 256, 0, stream>>>(hbuf, gid, w_atom, b_atom, hg, H);
  fix_max_kernel<<<blocks((long long)G * H), T, 0, stream>>>(hg, G, H);

  // ---- predictor MLP ----
  predictor_kernel<<<G, 256, 0, stream>>>(hg, P1, pb1, P2, pb2, pred, H);
}
